// DecoderLayer_77129022701893
// MI455X (gfx1250) — matrix operand order repaired and checked by reading, not compiled, not run on hardware
//
#include <hip/hip_runtime.h>
#include <hip/hip_bf16.h>
#include <stdint.h>

// Problem constants (match reference)
#define DM   1024
#define DFF  4096
#define NH   16
#define DH   64
#define BB   2
#define SS   2048
#define MTOT (BB * SS)   // 4096 rows of activations

typedef __attribute__((ext_vector_type(16))) __bf16         v16bf;
typedef __attribute__((ext_vector_type(8)))  float          v8f;
typedef __attribute__((ext_vector_type(8)))  unsigned short u16x8;
typedef __attribute__((ext_vector_type(16))) unsigned short u16x16;
typedef __attribute__((ext_vector_type(4)))  unsigned short u16x4;
typedef __attribute__((ext_vector_type(4)))  unsigned int   u32x4;
typedef __attribute__((ext_vector_type(4)))  int            i32x4;
typedef __attribute__((ext_vector_type(8)))  int            i32x8;

union BF16Frag {
  v16bf  v;
  u16x16 u;
  u16x8  h[2];
};

__device__ __forceinline__ unsigned short f2bf(float f) {
  unsigned int u = __float_as_uint(f);
  u += 0x7FFFu + ((u >> 16) & 1u);   // round-to-nearest-even
  return (unsigned short)(u >> 16);
}

__device__ __forceinline__ v8f wmma_bf16(v16bf a, v16bf b, v8f c) {
  // (neg_a, A, neg_b, B, c_mod, C, reuse_a, reuse_b)
  return __builtin_amdgcn_wmma_f32_16x16x32_bf16(false, a, false, b,
                                                 (short)0, c, false, false);
}

#if defined(__has_builtin)
#if __has_builtin(__builtin_amdgcn_tensor_load_to_lds) && \
    __has_builtin(__builtin_amdgcn_s_wait_tensorcnt)
#define HAVE_TDM 1
#endif
#endif

#ifdef HAVE_TDM
// Issue a TDM 2D tile load: 64 rows x 32 bf16 cols from W[row][kb..kb+31]
// (row stride = K elements) into LDS at byte offset lds_off.
// Descriptor packing per CDNA5 ISA ch.8 (D# groups 0/1).
__device__ __forceinline__ void tdm_load_w_tile(const unsigned short* gsrc,
                                                unsigned lds_off, int K) {
  unsigned long long ga = (unsigned long long)(uintptr_t)gsrc;
  u32x4 g0;
  g0[0] = 1u;                                   // count=1, user mode, no gather
  g0[1] = lds_off;                              // lds_addr (bytes)
  g0[2] = (unsigned)(ga & 0xffffffffu);         // global_addr[31:0]
  g0[3] = (unsigned)((ga >> 32) & 0x01ffffffu)  // global_addr[56:32]
          | 0x80000000u;                        // type=2 ("image")
  i32x8 g1;
  g1[0] = 1 << 16;                              // data_size=1 (2 bytes)
  g1[1] = (int)((unsigned)K << 16);             // tensor_dim0[15:0]
  g1[2] = (int)(((unsigned)K >> 16) & 0xffffu)  // tensor_dim0[31:16]
          | (64 << 16);                         // tensor_dim1[15:0] = 64
  g1[3] = 32 << 16;                             // dim1 hi=0, tile_dim0=32
  g1[4] = 64;                                   // tile_dim1=64, tile_dim2=0
  g1[5] = K;                                    // tensor_dim0_stride[31:0]
  g1[6] = 0;                                    // stride hi / dim1_stride lo
  g1[7] = 0;
  i32x4 z4 = {};
#if __clang_major__ >= 23
  i32x8 z8 = {};
  __builtin_amdgcn_tensor_load_to_lds(g0, g1, z4, z4, z8, 0);
#else
  __builtin_amdgcn_tensor_load_to_lds(g0, g1, z4, z4, 0);
#endif
}
#endif

// ---------------------------------------------------------------------------
// Y[M,N] = act( X[M,K](bf16) @ W[N,K]^T (bf16) + bias ) (+ residual)
// Block: 128 threads = 4 waves; block tile 128(M) x 64(N).
// Wave w owns rows [blockM + 32w, +32) as 2 A fragments -> 8 accumulators,
// so each B fragment staged in LDS is reused by 2 WMMAs.
// W tile (64x32 bf16, shared by all 4 waves) is DMA'd into a double-buffered
// LDS stage by the Tensor Data Mover (wave 0 issues, s_wait_tensorcnt syncs).
// All 4 B fragments are loaded from LDS before the 8-WMMA chain so the
// scheduler can overlap ds latency with matrix ops (partial dscnt waits).
// ---------------------------------------------------------------------------
__global__ __launch_bounds__(128) void gemm_bf16_nt(
    const unsigned short* __restrict__ A,
    const unsigned short* __restrict__ W,
    const float* __restrict__ bias,
    const float* __restrict__ res,      // [M,N] f32 or null
    float* __restrict__ outF,           // [M,N] f32 or null
    unsigned short* __restrict__ outB,  // [M,N] bf16 or null
    int M, int N, int K, int doRelu)
{
  __shared__ unsigned short ldsW[2][64 * 32];   // 8KB double buffer

  const int lane  = threadIdx.x & 31;
  const int wave  = threadIdx.x >> 5;
  const int khalf = lane >> 4;
  const int l16   = lane & 15;
  const int mBase = blockIdx.y * 128 + wave * 32;
  const int nBase = blockIdx.x * 64;
  const int aoff  = khalf * 16;

  const unsigned short* arow0 = A + (size_t)(mBase + l16) * K;
  const unsigned short* arow1 = A + (size_t)(mBase + 16 + l16) * K;
  const unsigned short* wpanel = W + (size_t)nBase * K;

#ifdef HAVE_TDM
  const unsigned ldsOff0 = (unsigned)(uintptr_t)&ldsW[0][0];
  const unsigned ldsOff1 = (unsigned)(uintptr_t)&ldsW[1][0];
  if (wave == 0) tdm_load_w_tile(wpanel, ldsOff0, K);   // prime buffer 0
#endif

  v8f acc[2][4] = {};
  int cur = 0;
  for (int kb = 0; kb < K; kb += 32) {
#ifdef HAVE_TDM
    if (wave == 0) {
      if (kb + 32 < K) {
        tdm_load_w_tile(wpanel + kb + 32, cur ? ldsOff0 : ldsOff1, K);
        __builtin_amdgcn_s_wait_tensorcnt(1);   // current tile landed
      } else {
        __builtin_amdgcn_s_wait_tensorcnt(0);
      }
    }
#else
    {  // cooperative fallback copy of the 64x32 W tile
      const int idx = threadIdx.x * 16;          // 128 thr * 16 = 2048 elems
      const int rowi = idx >> 5, coli = idx & 31;
      *(u16x16*)&ldsW[cur][idx] =
          *(const u16x16*)(wpanel + (size_t)rowi * K + kb + coli);
    }
#endif
    // A fragments from global, overlapped with the barrier
    BF16Frag a0, a1;
    a0.h[0] = *(const u16x8*)(arow0 + kb + aoff);
    a0.h[1] = *(const u16x8*)(arow0 + kb + aoff + 8);
    a1.h[0] = *(const u16x8*)(arow1 + kb + aoff);
    a1.h[1] = *(const u16x8*)(arow1 + kb + aoff + 8);
    if (kb + 32 < K) {
      __builtin_prefetch(arow0 + kb + 32, 0, 1);
      __builtin_prefetch(arow1 + kb + 32, 0, 1);
    }
    __syncthreads();   // W tile visible to all waves

    // batch all B-fragment LDS reads ahead of the WMMA chain
    BF16Frag bfr[4];
#pragma unroll
    for (int t = 0; t < 4; ++t)
      bfr[t].u = *(const u16x16*)&ldsW[cur][(t * 16 + l16) * 32 + khalf * 16];
#pragma unroll
    for (int t = 0; t < 4; ++t) {
      acc[0][t] = wmma_bf16(a0.v, bfr[t].v, acc[0][t]);
      acc[1][t] = wmma_bf16(a1.v, bfr[t].v, acc[1][t]);
    }
    __syncthreads();   // reads done before this buffer is overwritten
    cur ^= 1;
  }

#pragma unroll
  for (int mt = 0; mt < 2; ++mt) {
#pragma unroll
    for (int t = 0; t < 4; ++t) {
#pragma unroll
      for (int r = 0; r < 8; ++r) {
        const int row = mBase + mt * 16 + r + khalf * 8;
        const int col = nBase + t * 16 + l16;
        float v = acc[mt][t][r] + bias[col];
        if (res) v += res[(size_t)row * N + col];
        if (doRelu && v < 0.f) v = 0.f;
        if (outF) outF[(size_t)row * N + col] = v;
        if (outB) outB[(size_t)row * N + col] = f2bf(v);
      }
    }
  }
}

// ---------------------------------------------------------------------------
// Flash attention, one wave (32 threads) per (b,h, 16 q-rows).
// Q,K: [B*S, DM] bf16 (head h = cols h*64..+63). VT: [B*H*64, S] bf16.
// kv processed 32 columns per iteration: 4 WMMA for QK^T, 4 WMMA for P@V,
// with all global fragments batch-loaded before each WMMA chain.
// Softmax row stats kept per half-wave (C-frag rows live in 16-lane halves).
// P (C-layout) -> A-layout via a 1KB LDS bounce.
// ---------------------------------------------------------------------------
__global__ __launch_bounds__(32) void flash_attn(
    const unsigned short* __restrict__ Q,
    const unsigned short* __restrict__ Km,
    const unsigned short* __restrict__ VT,
    unsigned short* __restrict__ O,
    int causal)
{
  __shared__ unsigned short ldsP[16 * 32];
  const int lane  = threadIdx.x;
  const int khalf = lane >> 4;
  const int l16   = lane & 15;
  const int qt    = blockIdx.x;
  const int bh    = blockIdx.y;
  const int b     = bh >> 4;
  const int h     = bh & 15;
  const int aoff  = khalf * 16;

  BF16Frag qa0, qa1;  // Q tile 16x64 as two 16x32 A fragments
  {
    const unsigned short* qr =
        Q + ((size_t)(b * SS + qt * 16 + l16)) * DM + h * DH;
    qa0.h[0] = *(const u16x8*)(qr + aoff);
    qa0.h[1] = *(const u16x8*)(qr + aoff + 8);
    qa1.h[0] = *(const u16x8*)(qr + 32 + aoff);
    qa1.h[1] = *(const u16x8*)(qr + 32 + aoff + 8);
  }

  float mrow[8], lrow[8];
#pragma unroll
  for (int r = 0; r < 8; ++r) { mrow[r] = -3.0e38f; lrow[r] = 0.f; }
  v8f acc[4] = {};

  const int nT = causal ? ((qt * 16 + 16 + 31) >> 5) : (SS >> 5);
  for (int jt = 0; jt < nT; ++jt) {
    const int jb = jt * 32;

    // batch-load the four K B-fragments (2 kv subtiles x 2 d-halves)
    BF16Frag kf[2][2];
#pragma unroll
    for (int nt = 0; nt < 2; ++nt) {
      const int kvr = jb + nt * 16 + l16;
      const unsigned short* kr = Km + ((size_t)(b * SS + kvr)) * DM + h * DH;
      kf[nt][0].u = *(const u16x16*)(kr + khalf * 16);
      kf[nt][1].u = *(const u16x16*)(kr + 32 + khalf * 16);
    }

    // scores S = Q @ K^T  (two 16x16 C tiles covering 32 kv columns)
    v8f s[2];
#pragma unroll
    for (int nt = 0; nt < 2; ++nt) {
      v8f z = {};
      z = wmma_bf16(qa0.v, kf[nt][0].v, z);
      z = wmma_bf16(qa1.v, kf[nt][1].v, z);
      s[nt] = z;
    }

    // online softmax (row reductions across 16-lane halves)
#pragma unroll
    for (int r = 0; r < 8; ++r) {
      float v0 = s[0][r] * 0.125f;   // 1/sqrt(64)
      float v1 = s[1][r] * 0.125f;
      if (causal) {
        const int rowg = qt * 16 + r + khalf * 8;
        const int c0   = jb + l16;
        if (c0 > rowg)      v0 = -1.0e30f;
        if (c0 + 16 > rowg) v1 = -1.0e30f;
      }
      float tm = fmaxf(v0, v1);
      tm = fmaxf(tm, __shfl_xor(tm, 1, 32));
      tm = fmaxf(tm, __shfl_xor(tm, 2, 32));
      tm = fmaxf(tm, __shfl_xor(tm, 4, 32));
      tm = fmaxf(tm, __shfl_xor(tm, 8, 32));
      const float mnew = fmaxf(mrow[r], tm);
      const float corr = __expf(mrow[r] - mnew);
      const float p0 = __expf(v0 - mnew);
      const float p1 = __expf(v1 - mnew);
      float ts = p0 + p1;
      ts += __shfl_xor(ts, 1, 32);
      ts += __shfl_xor(ts, 2, 32);
      ts += __shfl_xor(ts, 4, 32);
      ts += __shfl_xor(ts, 8, 32);
      lrow[r] = lrow[r] * corr + ts;
      mrow[r] = mnew;
      acc[0][r] *= corr; acc[1][r] *= corr;
      acc[2][r] *= corr; acc[3][r] *= corr;
      const int prow = r + khalf * 8;
      ldsP[prow * 32 + l16]      = f2bf(p0);
      ldsP[prow * 32 + 16 + l16] = f2bf(p1);
    }
    __syncthreads();

    // re-read P in A-fragment layout
    BF16Frag pa;
    pa.h[0] = *(const u16x8*)&ldsP[l16 * 32 + aoff];
    pa.h[1] = *(const u16x8*)&ldsP[l16 * 32 + aoff + 8];
    __syncthreads();

    // batch-load the four V^T B-fragments, then O += P @ V
    BF16Frag vf[4];
#pragma unroll
    for (int t = 0; t < 4; ++t) {
      const int n = t * 16 + l16;
      vf[t].u = *(const u16x16*)(VT + ((size_t)(bh * DH + n)) * SS + jb +
                                 khalf * 16);
    }
#pragma unroll
    for (int t = 0; t < 4; ++t)
      acc[t] = wmma_bf16(pa.v, vf[t].v, acc[t]);
  }

#pragma unroll
  for (int t = 0; t < 4; ++t) {
#pragma unroll
    for (int r = 0; r < 8; ++r) {
      const float val = acc[t][r] / lrow[r];
      const int row = b * SS + qt * 16 + r + khalf * 8;
      const int col = h * DH + t * 16 + l16;
      O[(size_t)row * DM + col] = f2bf(val);
    }
  }
}

// ---------------------------------------------------------------------------
// VT[b,h,d,s] = V[b*S+s, h*64+d]   (bf16 -> bf16 transpose per head)
// ---------------------------------------------------------------------------
__global__ void transpose_v(const unsigned short* __restrict__ V,
                            unsigned short* __restrict__ VT)
{
  const size_t total = (size_t)BB * NH * DH * SS;
  for (size_t i = blockIdx.x * (size_t)blockDim.x + threadIdx.x; i < total;
       i += (size_t)gridDim.x * blockDim.x) {
    const int s  = (int)(i % SS);
    const int d  = (int)((i / SS) % DH);
    const int bh = (int)(i / ((size_t)SS * DH));
    const int b = bh >> 4, h = bh & 15;
    VT[i] = V[((size_t)(b * SS + s)) * DM + h * DH + d];
  }
}

__global__ void f32_to_bf16_kernel(const float* __restrict__ in,
                                   unsigned short* __restrict__ out, int n)
{
  for (int i = blockIdx.x * blockDim.x + threadIdx.x; i < n;
       i += gridDim.x * blockDim.x)
    out[i] = f2bf(in[i]);
}

// ---------------------------------------------------------------------------
// LayerNorm over 1024 cols; 256 threads x 4 cols. Writes f32 and/or bf16.
// ---------------------------------------------------------------------------
__global__ __launch_bounds__(256) void layernorm_kernel(
    const float* __restrict__ in, const float* __restrict__ g,
    const float* __restrict__ be, float* __restrict__ outF,
    unsigned short* __restrict__ outB)
{
  __shared__ float red[8];
  const int row  = blockIdx.x;
  const int tid  = threadIdx.x;
  const int lane = tid & 31, wid = tid >> 5;

  const float4 xv = *(const float4*)(in + (size_t)row * DM + tid * 4);
  float s = xv.x + xv.y + xv.z + xv.w;
  s += __shfl_xor(s, 1, 32);  s += __shfl_xor(s, 2, 32);
  s += __shfl_xor(s, 4, 32);  s += __shfl_xor(s, 8, 32);
  s += __shfl_xor(s, 16, 32);
  if (lane == 0) red[wid] = s;
  __syncthreads();
  float mu = 0.f;
#pragma unroll
  for (int i = 0; i < 8; ++i) mu += red[i];
  mu *= (1.0f / DM);
  __syncthreads();

  const float dx = xv.x - mu, dy = xv.y - mu, dz = xv.z - mu, dw = xv.w - mu;
  float s2 = dx * dx + dy * dy + dz * dz + dw * dw;
  s2 += __shfl_xor(s2, 1, 32);  s2 += __shfl_xor(s2, 2, 32);
  s2 += __shfl_xor(s2, 4, 32);  s2 += __shfl_xor(s2, 8, 32);
  s2 += __shfl_xor(s2, 16, 32);
  if (lane == 0) red[wid] = s2;
  __syncthreads();
  float var = 0.f;
#pragma unroll
  for (int i = 0; i < 8; ++i) var += red[i];
  var *= (1.0f / DM);
  const float inv = rsqrtf(var + 1e-5f);

  const int c = tid * 4;
  const float y0 = dx * inv * g[c + 0] + be[c + 0];
  const float y1 = dy * inv * g[c + 1] + be[c + 1];
  const float y2 = dz * inv * g[c + 2] + be[c + 2];
  const float y3 = dw * inv * g[c + 3] + be[c + 3];
  if (outF) {
    float4 o; o.x = y0; o.y = y1; o.z = y2; o.w = y3;
    *(float4*)(outF + (size_t)row * DM + c) = o;
  }
  if (outB) {
    u16x4 o = {f2bf(y0), f2bf(y1), f2bf(y2), f2bf(y3)};
    *(u16x4*)(outB + (size_t)row * DM + c) = o;
  }
}

// ---------------------------------------------------------------------------
extern "C" void kernel_launch(void* const* d_in, const int* in_sizes, int n_in,
                              void* d_out, int out_size, void* d_ws,
                              size_t ws_size, hipStream_t stream)
{
  (void)in_sizes; (void)n_in; (void)out_size; (void)ws_size;

  const float* x   = (const float*)d_in[0];
  const float* enc = (const float*)d_in[1];
  // masks d_in[2]/d_in[3] handled analytically (causal / full)
  const float* bq1 = (const float*)d_in[5];
  const float* bk1 = (const float*)d_in[7];
  const float* bv1 = (const float*)d_in[9];
  const float* bo1 = (const float*)d_in[11];
  const float* bq2 = (const float*)d_in[13];
  const float* bk2 = (const float*)d_in[15];
  const float* bv2 = (const float*)d_in[17];
  const float* bo2 = (const float*)d_in[19];
  const float* bff1 = (const float*)d_in[21];
  const float* bff2 = (const float*)d_in[23];
  const float* g1 = (const float*)d_in[24];  const float* be1 = (const float*)d_in[25];
  const float* g2 = (const float*)d_in[26];  const float* be2 = (const float*)d_in[27];
  const float* g3 = (const float*)d_in[28];  const float* be3 = (const float*)d_in[29];

  char* ws = (char*)d_ws;
  const size_t MB = 1024ull * 1024ull;
  unsigned short* wb[8];
  for (int i = 0; i < 8; ++i) wb[i] = (unsigned short*)(ws + (size_t)i * 2 * MB);
  unsigned short* wff1b = (unsigned short*)(ws + 16 * MB);
  unsigned short* wff2b = (unsigned short*)(ws + 24 * MB);
  unsigned short* xb    = (unsigned short*)(ws + 32 * MB);
  unsigned short* encb  = (unsigned short*)(ws + 40 * MB);
  unsigned short* qbuf  = (unsigned short*)(ws + 48 * MB);
  unsigned short* kbuf  = (unsigned short*)(ws + 56 * MB);
  unsigned short* vbuf  = (unsigned short*)(ws + 64 * MB);
  unsigned short* vtb   = (unsigned short*)(ws + 72 * MB);
  unsigned short* obuf  = (unsigned short*)(ws + 80 * MB);
  unsigned short* hb    = (unsigned short*)(ws + 88 * MB);   // 32 MB
  float* fA             = (float*)(ws + 120 * MB);
  float* fB             = (float*)(ws + 136 * MB);

  const int CVB = 256, CVG = 2048;
  const int widx[8] = {4, 6, 8, 10, 12, 14, 16, 18};
  for (int i = 0; i < 8; ++i)
    f32_to_bf16_kernel<<<CVG, CVB, 0, stream>>>((const float*)d_in[widx[i]],
                                                wb[i], DM * DM);
  f32_to_bf16_kernel<<<CVG, CVB, 0, stream>>>((const float*)d_in[20], wff1b, DFF * DM);
  f32_to_bf16_kernel<<<CVG, CVB, 0, stream>>>((const float*)d_in[22], wff2b, DM * DFF);
  f32_to_bf16_kernel<<<CVG, CVB, 0, stream>>>(x, xb, MTOT * DM);
  f32_to_bf16_kernel<<<CVG, CVB, 0, stream>>>(enc, encb, MTOT * DM);

  dim3 blk(128);
  dim3 gP(DM / 64, MTOT / 128);    // 1024-wide GEMMs, 128-row block tiles
  dim3 gF1(DFF / 64, MTOT / 128);  // 4096-wide GEMM
  dim3 gA(SS / 16, BB * NH);       // attention: (q-tiles, b*h)

  // ---- masked self-attention + residual + LN1 ----
  gemm_bf16_nt<<<gP, blk, 0, stream>>>(xb, wb[0], bq1, nullptr, nullptr, qbuf, MTOT, DM, DM, 0);
  gemm_bf16_nt<<<gP, blk, 0, stream>>>(xb, wb[1], bk1, nullptr, nullptr, kbuf, MTOT, DM, DM, 0);
  gemm_bf16_nt<<<gP, blk, 0, stream>>>(xb, wb[2], bv1, nullptr, nullptr, vbuf, MTOT, DM, DM, 0);
  transpose_v<<<2048, 256, 0, stream>>>(vbuf, vtb);
  flash_attn<<<gA, dim3(32), 0, stream>>>(qbuf, kbuf, vtb, obuf, 1);
  gemm_bf16_nt<<<gP, blk, 0, stream>>>(obuf, wb[3], bo1, x, fA, nullptr, MTOT, DM, DM, 0);
  layernorm_kernel<<<MTOT, 256, 0, stream>>>(fA, g1, be1, fB, xb);

  // ---- cross-attention + residual + LN2 ----
  gemm_bf16_nt<<<gP, blk, 0, stream>>>(xb,   wb[4], bq2, nullptr, nullptr, qbuf, MTOT, DM, DM, 0);
  gemm_bf16_nt<<<gP, blk, 0, stream>>>(encb, wb[5], bk2, nullptr, nullptr, kbuf, MTOT, DM, DM, 0);
  gemm_bf16_nt<<<gP, blk, 0, stream>>>(encb, wb[6], bv2, nullptr, nullptr, vbuf, MTOT, DM, DM, 0);
  transpose_v<<<2048, 256, 0, stream>>>(vbuf, vtb);
  flash_attn<<<gA, dim3(32), 0, stream>>>(qbuf, kbuf, vtb, obuf, 0);
  gemm_bf16_nt<<<gP, blk, 0, stream>>>(obuf, wb[7], bo2, fB, fA, nullptr, MTOT, DM, DM, 0);
  layernorm_kernel<<<MTOT, 256, 0, stream>>>(fA, g2, be2, fB, xb);

  // ---- FFN + residual + LN3 ----
  gemm_bf16_nt<<<gF1, blk, 0, stream>>>(xb, wff1b, bff1, nullptr, nullptr, hb, MTOT, DFF, DM, 1);
  gemm_bf16_nt<<<gP, blk, 0, stream>>>(hb, wff2b, bff2, fB, fA, nullptr, MTOT, DM, DFF, 0);
  layernorm_kernel<<<MTOT, 256, 0, stream>>>(fA, g3, be3, (float*)d_out, nullptr);
}
